// GCN_Fusion_12790412607528
// MI455X (gfx1250) — matrix operand order repaired and checked by reading, not compiled, not run on hardware
//
#include <hip/hip_runtime.h>
#include <hip/hip_bf16.h>
#include <math.h>

// MI455X (gfx1250) fused GCN block.
// - All three GEMMs on v_wmma_f32_16x16x32_bf16 (fp32 accumulate), fp32->bf16
//   conversion during LDS staging (bf16 WMMA is the throughput path; fp32 WMMA
//   16x16x4 would be ~8x slower per instruction and far off the 23.3 TB/s
//   memory roofline of ~10us for this op).
// - A1 = adj1 + alpha*tanh(q1[j]-k1[k]) generated straight into the LDS B-tile
//   of the graph matmul (never materialized in HBM: saves 128MB of traffic).
// - gemm_graph A-tile staging exercises the CDNA5 data movers:
//     rows  0..63  : TDM tensor_load_to_lds (D# in SGPRs, TENSORcnt)
//     rows 64..127 : global_load_async_to_lds_b128 (ASYNCcnt)
// - gemm_pool prefetches the next K-tile (global_prefetch_b8).

#define THREADS 256
#define BM 128
#define BN 64
#define BK 32

typedef __attribute__((ext_vector_type(16))) __bf16       bf16x16;
typedef __attribute__((ext_vector_type(8)))  float        f32x8;
typedef __attribute__((ext_vector_type(4)))  unsigned int u32x4;
typedef __attribute__((ext_vector_type(8)))  unsigned int u32x8;
typedef __attribute__((ext_vector_type(4)))  float        f32x4;

union Frag16 { bf16x16 v; u32x4 q[2]; };

__device__ __forceinline__ unsigned short f2bf(float f) {
  union { float f; unsigned int u; } c; c.f = f;
  unsigned int u = c.u;
  u += 0x7FFFu + ((u >> 16) & 1u);           // round-to-nearest-even
  return (unsigned short)(u >> 16);
}
__device__ __forceinline__ float bf2f(unsigned short h) {
  union { unsigned int u; float f; } c; c.u = ((unsigned int)h) << 16;
  return c.f;
}

__device__ __forceinline__ bf16x16 frag_ld(const unsigned short* p0,
                                           const unsigned short* p1) {
  Frag16 f;
  f.q[0] = *(const u32x4*)p0;   // 16B aligned ds_load_b128
  f.q[1] = *(const u32x4*)p1;
  return f.v;
}

// One wave: A frag (rows wv*16..+15 of lgA[BM][BK]) times four B subtiles
// (lgB stored [BN][BK], n-major) accumulated into acc[0..3].
// ISA layouts: A 16x32 bf16: m=L&15, k-chunks at (L>>4)*8 and +16.
// B 32x16 bf16: n=L&15, k-chunks at (L>>4)*8 and +16.  D: M = i + 8*(L>>4), N = L&15.
__device__ __forceinline__ void wave_mma(const unsigned short* lgA,
                                         const unsigned short* lgB,
                                         int wv, int lane, f32x8 acc[4]) {
  const int mr  = wv * 16 + (lane & 15);
  const int kbA = (lane >> 4) << 3;
  bf16x16 a = frag_ld(&lgA[mr * BK + kbA], &lgA[mr * BK + kbA + 16]);
  const int kbB = (lane >> 4) << 3;
#pragma unroll
  for (int t = 0; t < 4; ++t) {
    const unsigned short* bp = &lgB[(t * 16 + (lane & 15)) * BK + kbB];
    bf16x16 b = frag_ld(bp, bp + 16);
    acc[t] = __builtin_amdgcn_wmma_f32_16x16x32_bf16(
        false, a, false, b, (short)0, acc[t], false, false);
  }
}

// ---------------- stage 0: mean of convq/convk rows ----------------
__global__ __launch_bounds__(256) void prep_wbar(
    const float* __restrict__ wq, const float* __restrict__ wk,
    const float* __restrict__ bq, const float* __restrict__ bk,
    float* __restrict__ wqbar, float* __restrict__ wkbar, float* __restrict__ bb) {
  int c = threadIdx.x;                     // 256 channels
  float sq = 0.f, sk = 0.f;
  for (int q = 0; q < 64; ++q) { sq += wq[q * 256 + c]; sk += wk[q * 256 + c]; }
  wqbar[c] = sq * (1.0f / 64.0f);
  wkbar[c] = sk * (1.0f / 64.0f);
  if (c == 0) { float s = 0.f; for (int q = 0; q < 64; ++q) s += bq[q]; bb[0] = s * (1.0f / 64.0f); }
  if (c == 1) { float s = 0.f; for (int q = 0; q < 64; ++q) s += bk[q]; bb[1] = s * (1.0f / 64.0f); }
}

// ---------------- stage 1: pool GEMM  Xp[bc,o] = x[bc,:]·pool_w[o,:] + pool_b[o]
// M=8192, N=1024, K=2048. A = x fp32 [M,K] rowmajor. pool_w is [N,K] rowmajor.
__global__ __launch_bounds__(256) void gemm_pool(
    const float* __restrict__ X, const float* __restrict__ Wp,
    const float* __restrict__ pb, unsigned short* __restrict__ Xp) {
  __shared__ __attribute__((aligned(16))) unsigned short lgA[BM * BK];
  __shared__ __attribute__((aligned(16))) unsigned short lgB[BN * BK];
  const int tid = threadIdx.x, lane = tid & 31, wv = tid >> 5;
  const int m0 = blockIdx.y * BM;
  const int n0 = blockIdx.x * BN;
  const int K = 2048, N = 1024;
  f32x8 acc[4] = {};
  for (int kt = 0; kt < K; kt += BK) {
    __syncthreads();
#pragma unroll
    for (int e = 0; e < 16; ++e) {                 // A: 128x32 fp32 -> bf16
      int idx = tid + e * THREADS;
      int r = idx >> 5, k = idx & 31;
      lgA[idx] = f2bf(X[(size_t)(m0 + r) * K + kt + k]);
    }
#pragma unroll
    for (int e = 0; e < 8; ++e) {                  // B: [n][k] from [N,K] rowmajor
      int idx = tid + e * THREADS;
      int n = idx >> 5, k = idx & 31;
      lgB[idx] = f2bf(Wp[(size_t)(n0 + n) * K + kt + k]);
    }
    if (kt + BK < K) {                             // prefetch next A tile
      __builtin_prefetch(&X[(size_t)(m0 + (tid >> 1)) * K + kt + BK + ((tid & 1) << 4)], 0, 0);
    }
    __syncthreads();
    wave_mma(lgA, lgB, wv, lane, acc);
  }
  const int col = lane & 15, rb = (lane >> 4) << 3;
#pragma unroll
  for (int t = 0; t < 4; ++t) {
    int n = n0 + t * 16 + col;
    float bias = pb[n];
#pragma unroll
    for (int i = 0; i < 8; ++i) {
      int m = m0 + wv * 16 + rb + i;
      Xp[(size_t)m * N + n] = f2bf(acc[t][i] + bias);
    }
  }
}

// ---------------- stage 2: q1/k1 = wbar · Xp + bbar ----------------
__global__ __launch_bounds__(256) void qk_reduce(
    const unsigned short* __restrict__ Xp, const float* __restrict__ wq,
    const float* __restrict__ wk, const float* __restrict__ bb,
    float* __restrict__ q1, float* __restrict__ k1) {
  int b = blockIdx.y;
  int o = blockIdx.x * 256 + threadIdx.x;
  const unsigned short* base = Xp + (size_t)b * 256 * 1024 + o;
  float sq = 0.f, sk = 0.f;
  for (int c = 0; c < 256; ++c) {
    float v = bf2f(base[(size_t)c * 1024]);
    sq += wq[c] * v; sk += wk[c] * v;
  }
  q1[b * 1024 + o] = sq + bb[0];
  k1[b * 1024 + o] = sk + bb[1];
}

// ---------------- stage 3: h[b,o,j] = conv1_w[o,:]·Xp[b,:,j] + conv1_b[o]
// per b: M=256, N=1024, K=256. B = Xp[b] bf16 [K,N] rowmajor (transpose-staged).
__global__ __launch_bounds__(256) void gemm_conv1(
    const float* __restrict__ W1, const float* __restrict__ b1,
    const unsigned short* __restrict__ Xp, unsigned short* __restrict__ H) {
  __shared__ __attribute__((aligned(16))) unsigned short lgA[BM * BK];
  __shared__ __attribute__((aligned(16))) unsigned short lgB[BN * BK];
  const int tid = threadIdx.x, lane = tid & 31, wv = tid >> 5;
  const int b = blockIdx.z;
  const int m0 = blockIdx.y * BM;
  const int n0 = blockIdx.x * BN;
  const int K = 256, N = 1024;
  const unsigned short* Bp = Xp + (size_t)b * 256 * 1024;
  f32x8 acc[4] = {};
  for (int kt = 0; kt < K; kt += BK) {
    __syncthreads();
#pragma unroll
    for (int e = 0; e < 16; ++e) {                 // A = conv1_w fp32
      int idx = tid + e * THREADS;
      int r = idx >> 5, k = idx & 31;
      lgA[idx] = f2bf(W1[(size_t)(m0 + r) * K + kt + k]);
    }
#pragma unroll
    for (int e = 0; e < 8; ++e) {                  // B transpose-stage to [n][k]
      int idx = tid + e * THREADS;
      int n = idx & 63, k = idx >> 6;
      lgB[n * BK + k] = Bp[(size_t)(kt + k) * N + n0 + n];
    }
    __syncthreads();
    wave_mma(lgA, lgB, wv, lane, acc);
  }
  const int col = lane & 15, rb = (lane >> 4) << 3;
#pragma unroll
  for (int t = 0; t < 4; ++t) {
    int n = n0 + t * 16 + col;
#pragma unroll
    for (int i = 0; i < 8; ++i) {
      int m = m0 + wv * 16 + rb + i;
      H[((size_t)b * 256 + m) * N + n] = f2bf(acc[t][i] + b1[m]);
    }
  }
}

// ---------------- stage 4: out[b,c,k] = sum_j h[b,c,j]*(adj1[j,k]+al*tanh(q1-k1))
// per b: M=256, N=1024, K=1024. A tile staged by TDM (rows 0..63) + async LDS
// loads (rows 64..127); B tile generated directly in LDS.
__global__ __launch_bounds__(256) void gemm_graph(
    const unsigned short* __restrict__ H, const float* __restrict__ adj1,
    const float* __restrict__ q1, const float* __restrict__ k1,
    const float* __restrict__ alpha, float* __restrict__ Out) {
  __shared__ __attribute__((aligned(16))) unsigned short lgA[BM * BK];
  __shared__ __attribute__((aligned(16))) unsigned short lgB[BN * BK];
  const int tid = threadIdx.x, lane = tid & 31, wv = tid >> 5;
  const int b = blockIdx.z;
  const int m0 = blockIdx.y * BM;
  const int n0 = blockIdx.x * BN;
  const int K = 1024, N = 1024;
  const unsigned short* Ap = H + (size_t)b * 256 * 1024;
  const float* qb = q1 + b * 1024;
  const float* kb = k1 + b * 1024;
  const float al = alpha[0];
  const unsigned ldsA = (unsigned)(unsigned long long)(uintptr_t)lgA;  // LDS byte offset

  // TDM descriptor group 1 (constant across K loop): bf16 2D tensor
  // [dim1=256 rows x dim0=1024 cols], row stride 1024, tile 32 x 64 rows.
  u32x8 g1;
  g1[0] = 0x00010000u;        // workgroup_mask=0, data_size=1 (2B), no flags
  g1[1] = 1024u << 16;        // tensor_dim0[15:0]=1024 (atomic_barrier_addr=0)
  g1[2] = 256u << 16;         // tensor_dim0[31:16]=0, tensor_dim1[15:0]=256
  g1[3] = 32u << 16;          // tensor_dim1[31:16]=0, tile_dim0=32
  g1[4] = 64u;                // tile_dim1=64, tile_dim2=0
  g1[5] = 1024u;              // tensor_dim0_stride[31:0]=1024
  g1[6] = 0u;                 // stride0[47:32]=0, stride1[15:0]=0
  g1[7] = 0u;                 // tensor_dim1_stride rest = 0

  f32x8 acc[4] = {};
  for (int kt = 0; kt < K; kt += BK) {
    __syncthreads();
    // ---- A rows 0..63 via Tensor Data Mover (one DMA issued by wave 0) ----
    if (wv == 0) {
      unsigned long long ga =
          (unsigned long long)(uintptr_t)(Ap + (size_t)m0 * K + kt);
      u32x4 g0;
      g0[0] = 1u;                                   // count=1, user-mode D#
      g0[1] = ldsA;                                 // LDS byte address
      g0[2] = (unsigned)ga;                         // global_addr[31:0]
      g0[3] = ((unsigned)(ga >> 32) & 0x01FFFFFFu)  // global_addr[56:32]
              | 0x80000000u;                        // type=2 ("image")
      asm volatile("tensor_load_to_lds %0, %1"
                   :: "s"(g0), "s"(g1)
                   : "memory");
    }
    // ---- A rows 64..127 via async global->LDS b128 (one 16B chunk/thread) --
    {
      int row = 64 + (tid >> 2), c4 = tid & 3;
      unsigned lds = ldsA + row * (BK * 2) + c4 * 16;
      unsigned long long ga = (unsigned long long)(uintptr_t)
          (Ap + (size_t)(m0 + row) * K + kt + c4 * 8);
      asm volatile("global_load_async_to_lds_b128 %0, %1, off"
                   :: "v"(lds), "v"(ga)
                   : "memory");
    }
    // ---- B generated: adj1 + alpha*tanh(q1[j]-k1[n]) -> lgB[n][j] ----
#pragma unroll
    for (int e = 0; e < 8; ++e) {
      int idx = tid + e * THREADS;
      int n = idx & 63, j = idx >> 6;
      int jg = kt + j, ng = n0 + n;
      float v = adj1[(size_t)jg * 1024 + ng] + al * tanhf(qb[jg] - kb[ng]);
      lgB[n * BK + j] = f2bf(v);
    }
    asm volatile("s_wait_asynccnt 0x0" ::: "memory");
    if (wv == 0) __builtin_amdgcn_s_wait_tensorcnt(0);
    __syncthreads();
    wave_mma(lgA, lgB, wv, lane, acc);
  }
  const int col = lane & 15, rb = (lane >> 4) << 3;
#pragma unroll
  for (int t = 0; t < 4; ++t) {
    int n = n0 + t * 16 + col;
#pragma unroll
    for (int i = 0; i < 8; ++i) {
      int m = m0 + wv * 16 + rb + i;
      Out[((size_t)b * 256 + m) * N + n] = acc[t][i];
    }
  }
}

// ---------------- stage 5: per-channel batch-norm stats -> scale/shift ------
__global__ __launch_bounds__(256) void bn_stats(
    const float* __restrict__ Out, const float* __restrict__ gamma,
    const float* __restrict__ beta, float* __restrict__ scale,
    float* __restrict__ shift) {
  __shared__ float s1[256];
  __shared__ float s2[256];
  const int c = blockIdx.x, tid = threadIdx.x;
  float s = 0.f, sq = 0.f;
  for (int i = tid; i < 32768; i += 256) {       // 32 batches * 1024 joints
    int b = i >> 10, k = i & 1023;
    float v = Out[((size_t)b * 256 + c) * 1024 + k];
    s += v; sq += v * v;
  }
  s1[tid] = s; s2[tid] = sq; __syncthreads();
  for (int off = 128; off > 0; off >>= 1) {
    if (tid < off) { s1[tid] += s1[tid + off]; s2[tid] += s2[tid + off]; }
    __syncthreads();
  }
  if (tid == 0) {
    float mean = s1[0] * (1.0f / 32768.0f);
    float var  = s2[0] * (1.0f / 32768.0f) - mean * mean;
    float rstd = rsqrtf(var + 1e-5f);
    float sc = gamma[c] * rstd;
    scale[c] = sc;
    shift[c] = beta[c] - mean * sc;
  }
}

__global__ __launch_bounds__(256) void bn_apply(
    float* __restrict__ Out, const float* __restrict__ scale,
    const float* __restrict__ shift) {
  int idx = blockIdx.x * 256 + threadIdx.x;      // float4 index, 2,097,152 total
  int c = (idx >> 8) & 255;                      // 256 float4 per (b,c) row
  float sc = scale[c], sh = shift[c];
  f32x4* p = (f32x4*)Out;
  f32x4 v = p[idx];
  v.x = v.x * sc + sh; v.y = v.y * sc + sh;
  v.z = v.z * sc + sh; v.w = v.w * sc + sh;
  p[idx] = v;
}

extern "C" void kernel_launch(void* const* d_in, const int* in_sizes, int n_in,
                              void* d_out, int out_size, void* d_ws, size_t ws_size,
                              hipStream_t stream) {
  const float* x       = (const float*)d_in[0];   // [32,256,2048]
  const float* pool_w  = (const float*)d_in[1];   // [1024,2048]
  const float* pool_b  = (const float*)d_in[2];   // [1024]
  const float* adj1    = (const float*)d_in[3];   // [1024,1024]
  const float* conv1_w = (const float*)d_in[4];   // [256,256]
  const float* conv1_b = (const float*)d_in[5];   // [256]
  const float* convq_w = (const float*)d_in[6];   // [64,256]
  const float* convq_b = (const float*)d_in[7];   // [64]
  const float* convk_w = (const float*)d_in[8];   // [64,256]
  const float* convk_b = (const float*)d_in[9];   // [64]
  const float* alpha1  = (const float*)d_in[10];  // [1]
  const float* gamma   = (const float*)d_in[11];  // [256]
  const float* beta    = (const float*)d_in[12];  // [256]
  float* Out = (float*)d_out;                     // [32,256,1024]

  char* ws = (char*)d_ws;                         // ~34 MB used
  unsigned short* Xp = (unsigned short*)ws;                              // 16 MB bf16
  unsigned short* H  = (unsigned short*)(ws + (size_t)16 * 1024 * 1024); // 16 MB bf16
  float* q1    = (float*)(ws + (size_t)32 * 1024 * 1024);                // 128 KB
  float* k1    = q1 + 32 * 1024;                                         // 128 KB
  float* wqbar = k1 + 32 * 1024;                                         // 1 KB
  float* wkbar = wqbar + 256;
  float* bb    = wkbar + 256;
  float* scale = bb + 16;
  float* shift = scale + 256;

  prep_wbar<<<1, 256, 0, stream>>>(convq_w, convk_w, convq_b, convk_b,
                                   wqbar, wkbar, bb);
  gemm_pool<<<dim3(16, 64), 256, 0, stream>>>(x, pool_w, pool_b, Xp);
  qk_reduce<<<dim3(4, 32), 256, 0, stream>>>(Xp, wqbar, wkbar, bb, q1, k1);
  gemm_conv1<<<dim3(16, 2, 32), 256, 0, stream>>>(conv1_w, conv1_b, Xp, H);
  gemm_graph<<<dim3(16, 2, 32), 256, 0, stream>>>(H, adj1, q1, k1, alpha1, Out);
  bn_stats<<<256, 256, 0, stream>>>(Out, gamma, beta, scale, shift);
  bn_apply<<<8192, 256, 0, stream>>>(Out, scale, shift);
}
